// CLR_Pytorch_13855564497650
// MI455X (gfx1250) — compile-verified, matrix operand order checked by reading, not run on hardware
//
#include <hip/hip_runtime.h>
#include <math.h>

// ---------------------------------------------------------------------------
// SimCLR-style contrastive loss, fused, for MI455X (gfx1250, wave32, WMMA).
//   loss = mean_i [ log(sum_{j!=i} exp(2*cos(x_i,x_j))) - 2*cos(x_i, x_{i^1}) ]
// Pipeline:
//   K1: row L2-normalize fp32 -> split into bf16 hi/lo (x = hi + lo)
//   K2: tiled Gram matrix via bf16x3 WMMA (hi*hi + hi*lo + lo*hi, fp32 acc),
//       fused exp-sum + target-logit capture, per-block partial loss
//   K3: deterministic final reduction of 128 partials -> scalar mean
// ---------------------------------------------------------------------------

typedef __attribute__((ext_vector_type(16))) __bf16 v16bf;
typedef __attribute__((ext_vector_type(8)))  float  v8f;

#define NROWS   4096
#define DIM     1024
#define BLK_M   32                 // rows of sim per workgroup
#define N_TILE  16                 // WMMA N
#define TILE_G  4                  // N-tiles sharing one set of A registers
#define WAVES   8                  // 256 threads = 8 wave32
#define ASTRIDE (DIM + 8)          // LDS row stride (elements): +16B pad kills bank conflicts

union U32B { v16bf v; uint4 q[2]; };   // 16 bf16 = one lane's WMMA A/B operand

// --------------------------- K1: normalize + split -------------------------
__global__ __launch_bounds__(256) void k_norm_split(
    const float* __restrict__ x,
    unsigned short* __restrict__ hi_, unsigned short* __restrict__ lo_)
{
  const int row  = blockIdx.x;
  const int t    = threadIdx.x;
  const int lane = t & 31, wave = t >> 5;

  const float4 v = ((const float4*)(x + (size_t)row * DIM))[t]; // 4 elems/thread
  float ss = v.x*v.x + v.y*v.y + v.z*v.z + v.w*v.w;
  #pragma unroll
  for (int m = 16; m >= 1; m >>= 1) ss += __shfl_xor(ss, m, 32);

  __shared__ float red[8];
  if (lane == 0) red[wave] = ss;
  __syncthreads();
  float tot = 0.f;
  #pragma unroll
  for (int w = 0; w < 8; ++w) tot += red[w];

  const float scale = 1.0f / fmaxf(sqrtf(tot), 1e-8f);   // EPS clamp as reference
  const float xn[4] = { v.x*scale, v.y*scale, v.z*scale, v.w*scale };

  union { __bf16 b[4]; uint2 u; } H, L;
  #pragma unroll
  for (int i = 0; i < 4; ++i) {
    const __bf16 h = (__bf16)xn[i];                // RNE truncation to bf16
    H.b[i] = h;
    L.b[i] = (__bf16)(xn[i] - (float)h);           // residual -> second bf16
  }
  ((uint2*)(hi_ + (size_t)row * DIM))[t] = H.u;
  ((uint2*)(lo_ + (size_t)row * DIM))[t] = L.u;
}

// ------------------- K2: fused Gram + log-sum-exp + target ------------------
__global__ __launch_bounds__(256) void k_sim_loss(
    const unsigned short* __restrict__ hi_, const unsigned short* __restrict__ lo_,
    float* __restrict__ partial)
{
  const __bf16* __restrict__ ghi = (const __bf16*)hi_;
  const __bf16* __restrict__ glo = (const __bf16*)lo_;

  extern __shared__ __bf16 smem[];
  __bf16* Ahi = smem;
  __bf16* Alo = smem + BLK_M * ASTRIDE;
  __shared__ float waveSum[WAVES][BLK_M];
  __shared__ float tgtLogit[BLK_M];

  const int t    = threadIdx.x;
  const int wave = t >> 5, lane = t & 31;
  const int half = lane >> 4, l16 = lane & 15;
  const int rowBase = blockIdx.x * BLK_M;

  // Stage 32 x 1024 A-tile (hi & lo) into LDS: 16B per thread-iteration.
  for (int i = t; i < BLK_M * (DIM / 8); i += 256) {
    const int r = i >> 7;            // 128 8-elem chunks per row
    const int c = (i & 127) * 8;
    *(uint4*)(Ahi + r*ASTRIDE + c) = *(const uint4*)(ghi + (size_t)(rowBase + r)*DIM + c);
    *(uint4*)(Alo + r*ASTRIDE + c) = *(const uint4*)(glo + (size_t)(rowBase + r)*DIM + c);
  }
  if (t < BLK_M) tgtLogit[t] = 0.f;
  __syncthreads();

  float s[16];                        // per-lane running exp-sums, 16 rows
  #pragma unroll
  for (int j = 0; j < 16; ++j) s[j] = 0.f;
  const v8f vzero = {};

  // Each wave walks 8 groups of TILE_G=4 column tiles (all 4096 cols covered).
  for (int grp = wave; grp < (NROWS / N_TILE) / TILE_G; grp += WAVES) {
    const int n0 = grp * TILE_G * N_TILE;
    v8f acc[TILE_G][2];
    #pragma unroll
    for (int g = 0; g < TILE_G; ++g) { acc[g][0] = vzero; acc[g][1] = vzero; }

    for (int kb = 0; kb < DIM; kb += 32) {
      const int k0 = kb + 8 * half;   // 16-bit A layout: lane-half picks K phase
      U32B aH[2], aL[2];
      #pragma unroll
      for (int sub = 0; sub < 2; ++sub) {          // two 16-row M subtiles
        const __bf16* pa = Ahi + (sub*16 + l16)*ASTRIDE + k0;
        aH[sub].q[0] = *(const uint4*)(pa);
        aH[sub].q[1] = *(const uint4*)(pa + 16);
        const __bf16* pl = Alo + (sub*16 + l16)*ASTRIDE + k0;
        aL[sub].q[0] = *(const uint4*)(pl);
        aL[sub].q[1] = *(const uint4*)(pl + 16);
      }
      #pragma unroll
      for (int g = 0; g < TILE_G; ++g) {
        const size_t coff = (size_t)(n0 + g*N_TILE + l16) * DIM + k0;
        U32B bH, bL;                                // B = rows of xn (L2-resident)
        bH.q[0] = *(const uint4*)(ghi + coff);
        bH.q[1] = *(const uint4*)(ghi + coff + 16);
        bL.q[0] = *(const uint4*)(glo + coff);
        bL.q[1] = *(const uint4*)(glo + coff + 16);
        #pragma unroll
        for (int sub = 0; sub < 2; ++sub) {
          // bf16x3: hi*hi + hi*lo + lo*hi  (lo*lo ~ 2^-34, dropped)
          acc[g][sub] = __builtin_amdgcn_wmma_f32_16x16x32_bf16(
              false, aH[sub].v, false, bH.v, (short)0, acc[g][sub], false, false);
          acc[g][sub] = __builtin_amdgcn_wmma_f32_16x16x32_bf16(
              false, aH[sub].v, false, bL.v, (short)0, acc[g][sub], false, false);
          acc[g][sub] = __builtin_amdgcn_wmma_f32_16x16x32_bf16(
              false, aL[sub].v, false, bH.v, (short)0, acc[g][sub], false, false);
        }
      }
    }

    // Epilogue: |logit| <= 2, so exp needs no max-shift. Diagonal masked out.
    #pragma unroll
    for (int g = 0; g < TILE_G; ++g) {
      const int gj = n0 + g*N_TILE + l16;
      #pragma unroll
      for (int sub = 0; sub < 2; ++sub) {
        #pragma unroll
        for (int r = 0; r < 8; ++r) {
          const int row = sub*16 + r + 8*half;  // C layout: VGPR r, lane-half -> M
          const int gi  = rowBase + row;
          const float logit = 2.0f * acc[g][sub][r];   // /TEMPERATURE(0.5)
          if (gj == (gi ^ 1)) tgtLogit[row] = logit;   // augmentation partner
          s[sub*8 + r] += (gj == gi) ? 0.0f : __expf(logit);
        }
      }
    }
  }

  // Reduce exp-sums across the 16 lanes of each half (same rows per half).
  #pragma unroll
  for (int j = 0; j < 16; ++j) {
    float v = s[j];
    #pragma unroll
    for (int m = 1; m <= 8; m <<= 1) v += __shfl_xor(v, m, 32);
    s[j] = v;
  }
  if (l16 == 0) {
    #pragma unroll
    for (int j = 0; j < 16; ++j) {
      const int row = (j >> 3)*16 + (j & 7) + 8*half;
      waveSum[wave][row] = s[j];
    }
  }
  __syncthreads();

  // Wave 0: deterministic fixed-order merge across waves, then block partial.
  if (wave == 0) {
    const int row = lane;                      // 32 rows, one per lane
    float tot = 0.f;
    #pragma unroll
    for (int w = 0; w < WAVES; ++w) tot += waveSum[w][row];
    float lossRow = logf(tot) - tgtLogit[row]; // -logp[i, target]
    #pragma unroll
    for (int m = 1; m <= 16; m <<= 1) lossRow += __shfl_xor(lossRow, m, 32);
    if (lane == 0) partial[blockIdx.x] = lossRow;
  }
}

// ------------------------- K3: final mean reduction ------------------------
__global__ __launch_bounds__(128) void k_reduce(
    const float* __restrict__ partial, float* __restrict__ out, int n)
{
  const int t = threadIdx.x;
  float v = (t < n) ? partial[t] : 0.f;
  #pragma unroll
  for (int m = 16; m >= 1; m >>= 1) v += __shfl_xor(v, m, 32);
  __shared__ float red[4];
  if ((t & 31) == 0) red[t >> 5] = v;
  __syncthreads();
  if (t == 0) out[0] = (red[0] + red[1] + red[2] + red[3]) / (float)NROWS;
}

// ------------------------------- launcher ----------------------------------
extern "C" void kernel_launch(void* const* d_in, const int* in_sizes, int n_in,
                              void* d_out, int out_size, void* d_ws, size_t ws_size,
                              hipStream_t stream)
{
  const float* image = (const float*)d_in[0];
  float* out = (float*)d_out;

  unsigned short* hi = (unsigned short*)d_ws;                 // 8 MB
  unsigned short* lo = hi + (size_t)NROWS * DIM;              // 8 MB
  float* partial     = (float*)(lo + (size_t)NROWS * DIM);    // 128 floats

  k_norm_split<<<NROWS, 256, 0, stream>>>(image, hi, lo);

  const int nblk = NROWS / BLK_M;                              // 128
  const size_t smem = (size_t)2 * BLK_M * ASTRIDE * sizeof(__bf16); // ~129 KB of 320 KB/WGP
  k_sim_loss<<<nblk, 256, smem, stream>>>(hi, lo, partial);

  k_reduce<<<1, 128, 0, stream>>>(partial, out, nblk);
}
